// MultiHeadAttention_28767690948966
// MI455X (gfx1250) — matrix operand order repaired and checked by reading, not compiled, not run on hardware
//
#include <hip/hip_runtime.h>
#include <hip/hip_bf16.h>

typedef __attribute__((ext_vector_type(16))) __bf16 v16bf;
typedef __attribute__((ext_vector_type(8)))  __bf16 v8bf;
typedef __attribute__((ext_vector_type(2)))  __bf16 v2bf;
typedef __attribute__((ext_vector_type(8)))  float  v8f;

#define BATCH  2
#define SEQ    2048
#define DMODEL 512
#define NHEAD  8
#define HDIM   64

// ---------------------------------------------------------------------------
// Fragment loaders for v_wmma_f32_16x16x32_bf16 (wave32 layouts, ISA 7.12.2)
// A 16x32: lanes 0-15 hold K{0..7,16..23}, lanes 16-31 hold K{8..15,24..31}
// B 32x16: lanes 0-15 hold K{0..7,16..23}, lanes 16-31 hold K{8..15,24..31}
// `row` points at the start of this lane's 32-element K-run (16B aligned).
// ---------------------------------------------------------------------------
__device__ __forceinline__ v16bf frag_a_ld(const __bf16* row, bool hi) {
  const v8bf lo = *(const v8bf*)(row + (hi ? 8 : 0));
  const v8bf h2 = *(const v8bf*)(row + (hi ? 24 : 16));
  return __builtin_shufflevector(lo, h2, 0,1,2,3,4,5,6,7,8,9,10,11,12,13,14,15);
}
__device__ __forceinline__ v16bf frag_b_ld(const __bf16* row, bool hi) {
  const v8bf lo = *(const v8bf*)(row + (hi ? 8 : 0));
  const v8bf h2 = *(const v8bf*)(row + (hi ? 24 : 16));
  return __builtin_shufflevector(lo, h2, 0,1,2,3,4,5,6,7,8,9,10,11,12,13,14,15);
}

// ---------------------------------------------------------------------------
// DPP16 butterfly reductions across 16-lane rows (pure VALU, no LDS).
// Steps: quad_perm swap1, quad_perm swap2, row_half_mirror, row_mirror.
// After each step, lanes in the reduced group hold identical partials, so
// mirror patterns are valid butterfly substitutes.
// ---------------------------------------------------------------------------
template <int CTRL>
__device__ __forceinline__ float dpp_mov(float x) {
  return __int_as_float(__builtin_amdgcn_update_dpp(
      0, __float_as_int(x), CTRL, 0xf, 0xf, true));
}
__device__ __forceinline__ float rowmax16(float x) {
  x = fmaxf(x, dpp_mov<0xB1>(x));   // quad_perm [1,0,3,2]
  x = fmaxf(x, dpp_mov<0x4E>(x));   // quad_perm [2,3,0,1]
  x = fmaxf(x, dpp_mov<0x141>(x));  // row_half_mirror
  x = fmaxf(x, dpp_mov<0x140>(x));  // row_mirror
  return x;
}
__device__ __forceinline__ float rowsum16(float x) {
  x += dpp_mov<0xB1>(x);
  x += dpp_mov<0x4E>(x);
  x += dpp_mov<0x141>(x);
  x += dpp_mov<0x140>(x);
  return x;
}

// ---------------------------------------------------------------------------
// Tiled GEMM: [4096 x 512] (fp32) @ [512 x 512] (fp32), out = (acc+bias)*scale
// MODE 0: out = bf16, scattered to [B, H, S, HD] head-split layout
// MODE 1: out = fp32, row-major [4096 x 512] (final projection -> d_out)
// Block 256 threads = 8 waves (2 x 4), block tile 128x128, wave tile 64x32.
// ---------------------------------------------------------------------------
template <int MODE>
__global__ __launch_bounds__(256)
void gemm512_bf16(const float* __restrict__ A, const float* __restrict__ W,
                  const float* __restrict__ bias, void* __restrict__ dstv,
                  float oscale)
{
  constexpr int LDT = 40;              // padded LDS row stride (bf16), 80B
  __shared__ __bf16 As[128 * LDT];     // A tile, row-major (m, k)
  __shared__ __bf16 Bs[128 * LDT];     // W tile, transposed (n, k)

  const int t    = threadIdx.x;
  const int lane = t & 31;
  const int wave = t >> 5;
  const int wm   = wave >> 2;          // 0..1  -> 64 rows
  const int wn   = wave & 3;           // 0..3  -> 32 cols
  const int bm   = blockIdx.y;         // 0..31
  const int bn   = blockIdx.x;         // 0..3
  const bool hi  = lane >= 16;
  const int l16  = lane & 15;

  // per-thread staging roles
  const int arow  = t >> 1;            // A: 0..127
  const int ahalf = t & 1;             //    16 floats each
  const int kk    = (t >> 4) * 2;      // W: k pair base, 0,2,..,30
  const int seg   = t & 15;            //    n block of 8

  const float* asrc = A + (size_t)(bm * 128 + arow) * DMODEL + ahalf * 16;
  const float* wsrc = W + (size_t)kk * DMODEL + bn * 128 + seg * 8;

  const v8f vz = {0.f,0.f,0.f,0.f,0.f,0.f,0.f,0.f};
  v8f acc[4][2];
#pragma unroll
  for (int mf = 0; mf < 4; ++mf)
#pragma unroll
    for (int nf = 0; nf < 2; ++nf) acc[mf][nf] = vz;

  for (int k0 = 0; k0 < DMODEL; k0 += 32) {
    if (k0 + 32 < DMODEL) {            // prefetch next K-step
      __builtin_prefetch(asrc + k0 + 32, 0, 1);
      __builtin_prefetch(wsrc + (size_t)32 * DMODEL, 0, 1);
    }
    { // stage A: 128 rows x 32 K, fp32 -> bf16, row-major
      const float* src = asrc + k0;
      __bf16* d = As + arow * LDT + ahalf * 16;
#pragma unroll
      for (int i = 0; i < 4; ++i) {
        const float4 f = ((const float4*)src)[i];
        d[i*4+0] = (__bf16)f.x; d[i*4+1] = (__bf16)f.y;
        d[i*4+2] = (__bf16)f.z; d[i*4+3] = (__bf16)f.w;
      }
    }
    { // stage W transposed (n, k): packed 2xbf16 stores along k
      const float* s0 = wsrc + (size_t)k0 * DMODEL;
      const float* s1 = s0 + DMODEL;
#pragma unroll
      for (int i = 0; i < 2; ++i) {
        const float4 f0 = ((const float4*)s0)[i];
        const float4 f1 = ((const float4*)s1)[i];
        const int nb = seg * 8 + i * 4;
        *(v2bf*)&Bs[(nb+0)*LDT + kk] = v2bf{(__bf16)f0.x, (__bf16)f1.x};
        *(v2bf*)&Bs[(nb+1)*LDT + kk] = v2bf{(__bf16)f0.y, (__bf16)f1.y};
        *(v2bf*)&Bs[(nb+2)*LDT + kk] = v2bf{(__bf16)f0.z, (__bf16)f1.z};
        *(v2bf*)&Bs[(nb+3)*LDT + kk] = v2bf{(__bf16)f0.w, (__bf16)f1.w};
      }
    }
    __syncthreads();

    v16bf bf[2];
#pragma unroll
    for (int nf = 0; nf < 2; ++nf)
      bf[nf] = frag_b_ld(Bs + (wn * 32 + nf * 16 + l16) * LDT, hi);
#pragma unroll
    for (int mf = 0; mf < 4; ++mf) {
      const v16bf af = frag_a_ld(As + (wm * 64 + mf * 16 + l16) * LDT, hi);
#pragma unroll
      for (int nf = 0; nf < 2; ++nf)
        acc[mf][nf] = __builtin_amdgcn_wmma_f32_16x16x32_bf16(
            false, af, false, bf[nf], (short)0, acc[mf][nf], false, false);
    }
    __syncthreads();
  }

  // epilogue: C layout -> lane holds column n=l16, rows r(+8 for hi half)
  const int mrow0 = bm * 128 + wm * 64 + (hi ? 8 : 0);
  const int ncol0 = bn * 128 + wn * 32;
#pragma unroll
  for (int nf = 0; nf < 2; ++nf) {
    const int gn = ncol0 + nf * 16 + l16;
    const float bv = bias[gn];
#pragma unroll
    for (int mf = 0; mf < 4; ++mf) {
#pragma unroll
      for (int r = 0; r < 8; ++r) {
        const int gm = mrow0 + mf * 16 + r;
        const float val = (acc[mf][nf][r] + bv) * oscale;
        if (MODE == 0) {
          const int b = gm >> 11, s = gm & (SEQ - 1);
          const int h = gn >> 6,  d = gn & (HDIM - 1);
          ((__bf16*)dstv)[((size_t)(b * NHEAD + h) * SEQ + s) * HDIM + d] =
              (__bf16)val;
        } else {
          ((float*)dstv)[(size_t)gm * DMODEL + gn] = val;
        }
      }
    }
  }
}

// ---------------------------------------------------------------------------
// Flash attention: grid (S/128, H, B), block 256 threads = 8 waves.
// Each wave owns 16 query rows; streams 64-key chunks with online softmax.
// Q is pre-scaled by 1/sqrt(HD) in the projection epilogue.
// Q/K/V are bf16 in [B, H, S, HD]; ctx written fp32 row-major [B*S, 512].
// ---------------------------------------------------------------------------
__global__ __launch_bounds__(256)
void flash_attn(const __bf16* __restrict__ Q, const __bf16* __restrict__ K,
                const __bf16* __restrict__ V, float* __restrict__ ctx)
{
  constexpr int LDK = 72;                 // padded stride (144B, 16B-aligned)
  __shared__ __bf16 Ks[64 * LDK];         // K chunk, (key, d)
  __shared__ __bf16 Vt[64 * LDK];         // V chunk transposed, (d, key)
  __shared__ __bf16 Ps[8][16 * LDK];      // wave-private P tiles (q, key)

  const int t    = threadIdx.x;
  const int lane = t & 31;
  const int wave = t >> 5;
  const bool hi  = lane >= 16;
  const int l16  = lane & 15;
  const int h = blockIdx.y, b = blockIdx.z;
  const size_t base = (size_t)(b * NHEAD + h) * SEQ * HDIM;
  const __bf16* Qb = Q + base;
  const __bf16* Kb = K + base;
  const __bf16* Vb = V + base;
  const int q0 = blockIdx.x * 128 + wave * 16;

  // staging roles
  const int krow  = t >> 2, kpart = t & 3;        // K: 16 bf16 each
  const int vkk   = (t >> 3) * 2, vpart = t & 7;  // V: key pair, 8 d each

  v16bf qf[2];
  {
    const __bf16* qrow = Qb + (size_t)(q0 + l16) * HDIM;
    qf[0] = frag_a_ld(qrow, hi);          // d 0..31
    qf[1] = frag_a_ld(qrow + 32, hi);     // d 32..63
  }

  const v8f vz = {0.f,0.f,0.f,0.f,0.f,0.f,0.f,0.f};
  v8f o[4] = {vz, vz, vz, vz};
  float rmax[8], rsum[8];
#pragma unroll
  for (int r = 0; r < 8; ++r) { rmax[r] = -3.0e38f; rsum[r] = 0.f; }

  for (int j = 0; j < SEQ / 64; ++j) {
    { // stage K chunk (direct copy, bf16)
      const __bf16* src = Kb + (size_t)(j * 64 + krow) * HDIM + kpart * 16;
      __bf16* d = Ks + krow * LDK + kpart * 16;
      *(v8bf*)(d)     = *(const v8bf*)(src);
      *(v8bf*)(d + 8) = *(const v8bf*)(src + 8);
    }
    { // stage V chunk transposed Vt[d][key]: packed 2xbf16 stores
      const __bf16* s0 = Vb + (size_t)(j * 64 + vkk) * HDIM + vpart * 8;
      const __bf16* s1 = s0 + HDIM;
      const v8bf a0 = *(const v8bf*)s0;
      const v8bf a1 = *(const v8bf*)s1;
#pragma unroll
      for (int i = 0; i < 8; ++i)
        *(v2bf*)&Vt[(vpart * 8 + i) * LDK + vkk] = v2bf{a0[i], a1[i]};
    }
    if (j + 1 < SEQ / 64) { // prefetch next chunk while this one computes
      __builtin_prefetch(Kb + (size_t)((j + 1) * 64 + krow) * HDIM + kpart * 16, 0, 1);
      __builtin_prefetch(Vb + (size_t)((j + 1) * 64 + vkk) * HDIM + vpart * 8, 0, 1);
    }
    __syncthreads();

    // scores S = (Q/8) K^T, 16 x 64 per wave (scale folded into Q)
    v8f s[4];
#pragma unroll
    for (int nf = 0; nf < 4; ++nf) {
      const __bf16* krw = Ks + (nf * 16 + l16) * LDK;
      const v16bf kb0 = frag_b_ld(krw, hi);
      const v16bf kb1 = frag_b_ld(krw + 32, hi);
      v8f a = vz;
      a = __builtin_amdgcn_wmma_f32_16x16x32_bf16(false, qf[0], false, kb0,
                                                  (short)0, a, false, false);
      a = __builtin_amdgcn_wmma_f32_16x16x32_bf16(false, qf[1], false, kb1,
                                                  (short)0, a, false, false);
      s[nf] = a;
    }

    // online softmax: DPP16 row reductions (no LDS traffic)
    float cmax[8], csum[8];
#pragma unroll
    for (int r = 0; r < 8; ++r) {
      cmax[r] = rowmax16(
          fmaxf(fmaxf(s[0][r], s[1][r]), fmaxf(s[2][r], s[3][r])));
      const float nm = fmaxf(rmax[r], cmax[r]);
      const float sc = __expf(rmax[r] - nm);
      rmax[r] = nm;
      rsum[r] *= sc;
#pragma unroll
      for (int df = 0; df < 4; ++df) o[df][r] *= sc;
      csum[r] = 0.f;
    }
#pragma unroll
    for (int nf = 0; nf < 4; ++nf)
#pragma unroll
      for (int r = 0; r < 8; ++r) {
        const float p = __expf(s[nf][r] - rmax[r]);
        s[nf][r] = p;
        csum[r] += p;
      }
#pragma unroll
    for (int r = 0; r < 8; ++r) rsum[r] += rowsum16(csum[r]);

    // P: C-fragment layout -> LDS -> A-fragment layout (wave-private)
    {
      __bf16* Pw = &Ps[wave][0];
      const int ro = hi ? 8 : 0;
#pragma unroll
      for (int nf = 0; nf < 4; ++nf)
#pragma unroll
        for (int r = 0; r < 8; ++r)
          Pw[(ro + r) * LDK + nf * 16 + l16] = (__bf16)s[nf][r];
    }
    asm volatile("s_wait_dscnt 0" ::: "memory"); // same-wave LDS RAW

    { // O += P @ V
      const __bf16* Pr = &Ps[wave][0] + l16 * LDK;
      const v16bf pa0 = frag_a_ld(Pr, hi);        // keys 0..31
      const v16bf pa1 = frag_a_ld(Pr + 32, hi);   // keys 32..63
#pragma unroll
      for (int df = 0; df < 4; ++df) {
        const __bf16* vrow = Vt + (df * 16 + l16) * LDK;
        const v16bf vb0 = frag_b_ld(vrow, hi);
        const v16bf vb1 = frag_b_ld(vrow + 32, hi);
        o[df] = __builtin_amdgcn_wmma_f32_16x16x32_bf16(
            false, pa0, false, vb0, (short)0, o[df], false, false);
        o[df] = __builtin_amdgcn_wmma_f32_16x16x32_bf16(
            false, pa1, false, vb1, (short)0, o[df], false, false);
      }
    }
    __syncthreads();
  }

  // normalize + merge heads into ctx [B*S, 512] fp32
#pragma unroll
  for (int r = 0; r < 8; ++r) rsum[r] = 1.0f / rsum[r];
  const int gq0 = q0 + (hi ? 8 : 0);
#pragma unroll
  for (int df = 0; df < 4; ++df) {
    const int col = h * HDIM + df * 16 + l16;
#pragma unroll
    for (int r = 0; r < 8; ++r) {
      const int gq = gq0 + r;
      ctx[(size_t)(b * SEQ + gq) * DMODEL + col] = o[df][r] * rsum[r];
    }
  }
}

// ---------------------------------------------------------------------------
// Host launcher
// ---------------------------------------------------------------------------
extern "C" void kernel_launch(void* const* d_in, const int* in_sizes, int n_in,
                              void* d_out, int out_size, void* d_ws, size_t ws_size,
                              hipStream_t stream) {
  (void)in_sizes; (void)n_in; (void)out_size; (void)ws_size;
  const float* q  = (const float*)d_in[0];
  const float* k  = (const float*)d_in[1];
  const float* v  = (const float*)d_in[2];
  const float* Wq = (const float*)d_in[3];
  const float* bq = (const float*)d_in[4];
  const float* Wk = (const float*)d_in[5];
  const float* bk = (const float*)d_in[6];
  const float* Wv = (const float*)d_in[7];
  const float* bv = (const float*)d_in[8];
  const float* Wo = (const float*)d_in[9];
  const float* bo = (const float*)d_in[10];

  char* ws = (char*)d_ws;
  const size_t qkv_bytes = (size_t)BATCH * NHEAD * SEQ * HDIM * sizeof(__bf16); // 4 MiB
  __bf16* qws = (__bf16*)(ws);
  __bf16* kws = (__bf16*)(ws + qkv_bytes);
  __bf16* vws = (__bf16*)(ws + 2 * qkv_bytes);
  float*  cws = (float*)(ws + 3 * qkv_bytes);   // ctx fp32, 8 MiB

  const dim3 ggrid(DMODEL / 128, (BATCH * SEQ) / 128);  // (4, 32)
  const dim3 gblk(256);

  const float qscale = 0.125f;   // 1/sqrt(HDIM), folded into Q projection

  gemm512_bf16<0><<<ggrid, gblk, 0, stream>>>(q, Wq, bq, (void*)qws, qscale);
  gemm512_bf16<0><<<ggrid, gblk, 0, stream>>>(k, Wk, bk, (void*)kws, 1.0f);
  gemm512_bf16<0><<<ggrid, gblk, 0, stream>>>(v, Wv, bv, (void*)vws, 1.0f);

  flash_attn<<<dim3(SEQ / 128, NHEAD, BATCH), dim3(256), 0, stream>>>(
      qws, kws, vws, cws);

  gemm512_bf16<1><<<ggrid, gblk, 0, stream>>>(cws, Wo, bo, d_out, 1.0f);
}
